// SelfAtten_58677843198654
// MI455X (gfx1250) — matrix operand order repaired and checked by reading, not compiled, not run on hardware
//
#include <hip/hip_runtime.h>
#include <stdint.h>

// ---------------------------------------------------------------------------
// Problem constants: B=8, S=4096, D=256 (from reference).
// ---------------------------------------------------------------------------
#define SEQ   4096
#define DIM   256
#define BATCH 8

typedef __attribute__((ext_vector_type(16))) __bf16          v16bf;
typedef __attribute__((ext_vector_type(8)))  __bf16          v8bf;
typedef __attribute__((ext_vector_type(8)))  float           v8f;
typedef __attribute__((ext_vector_type(16))) unsigned short  us16;
typedef __attribute__((ext_vector_type(8)))  unsigned short  us8;

union BfPack { us16 u; v16bf b; };

__device__ __forceinline__ v8f wmma_bf16(v16bf a, v16bf b, v8f c) {
  // D(16x16 f32) = A(16x32 bf16) * B(32x16 bf16) + C
  return __builtin_amdgcn_wmma_f32_16x16x32_bf16(false, a, false, b,
                                                 (short)0, c, false, false);
}

// ---------------------------------------------------------------------------
// Kernel 0a: convert src f32 -> bf16 row-major (done ONCE; pure bandwidth).
// ---------------------------------------------------------------------------
__global__ __launch_bounds__(256) void convert_src_kernel(
    const float* __restrict__ src, __bf16* __restrict__ Sb)
{
  const size_t i = ((size_t)blockIdx.x * 256 + threadIdx.x) * 8;
  v8f f = *(const v8f*)(src + i);
  *(v8bf*)(Sb + i) = __builtin_convertvector(f, v8bf);
}

// ---------------------------------------------------------------------------
// Kernel 0b: convert Wq/Wk/Wv f32 -> bf16 directly in packed B-operand layout:
//   Wp[m(3)][ctile(16)][dg(8)][lane(32)][j(16)], element =
//   W_m[e = ct*16 + (lane&15)][d = dg*32 + (lane>>4)*16 + j]
// ---------------------------------------------------------------------------
__global__ __launch_bounds__(256) void convert_w_kernel(
    const float* __restrict__ Wq, const float* __restrict__ Wk,
    const float* __restrict__ Wv, __bf16* __restrict__ Wp)
{
  const int idx = blockIdx.x * 256 + threadIdx.x;   // 0 .. 3*65536-1
  const int m    = idx >> 16;
  const int r    = idx & 65535;
  const int ct   = r >> 12;
  const int dg   = (r >> 9) & 7;
  const int lane = (r >> 4) & 31;
  const int j    = r & 15;
  const int e = ct * 16 + (lane & 15);
  const int d = dg * 32 + (lane >> 4) * 16 + j;
  const float* W = (m == 0) ? Wq : (m == 1) ? Wk : Wv;
  Wp[idx] = (__bf16)W[(size_t)e * DIM + d];
}

// ---------------------------------------------------------------------------
// Kernel 1: fused QKV projection, pure bf16 WMMA pipeline.
//   Q  -> row-major bf16 [B*S][D]
//   K  -> packed B-operand layout [b][ktile(256)][dg(8)][lane(32)][16] bf16
//   V  -> row-major f32  [B*S][D]
// ---------------------------------------------------------------------------
__global__ __launch_bounds__(256) void qkv_proj_kernel(
    const __bf16* __restrict__ Sb, const __bf16* __restrict__ Wp,
    __bf16* __restrict__ Qb, __bf16* __restrict__ Ktp, float* __restrict__ Vf)
{
  const int lane = threadIdx.x & 31;
  const int wrp  = threadIdx.x >> 5;
  const int wid  = blockIdx.x * 8 + wrp;     // 0..32767
  const int row_tile = wid >> 4;             // 0..2047  (16-row tiles of B*S)
  const int col_tile = wid & 15;             // 0..15    (16-col tiles of D)
  const int hl = lane >> 4;
  const int n  = lane & 15;
  const int e0 = col_tile * 16;

  const __bf16* arow = Sb + (size_t)(row_tile * 16 + n) * DIM;
  const __bf16* wq = Wp + ((size_t)(0 * 16 + col_tile) * 8) * 512 + lane * 16;
  const __bf16* wk = Wp + ((size_t)(1 * 16 + col_tile) * 8) * 512 + lane * 16;
  const __bf16* wv = Wp + ((size_t)(2 * 16 + col_tile) * 8) * 512 + lane * 16;

  v8f cq = {0.f,0.f,0.f,0.f,0.f,0.f,0.f,0.f};
  v8f ck = cq, cv = cq;

  #pragma unroll
  for (int dg = 0; dg < 8; ++dg) {
    const int d0 = dg * 32;
    us8 lo = *(const us8*)(arow + d0 + 16 * hl);
    us8 hi = *(const us8*)(arow + d0 + 16 * hl + 8);
    BfPack a;
    #pragma unroll
    for (int i = 0; i < 8; ++i) { a.u[i] = lo[i]; a.u[i + 8] = hi[i]; }
    BfPack bq, bk, bv;
    bq.u = *(const us16*)(wq + dg * 512);
    bk.u = *(const us16*)(wk + dg * 512);
    bv.u = *(const us16*)(wv + dg * 512);
    cq = wmma_bf16(a.b, bq.b, cq);
    ck = wmma_bf16(a.b, bk.b, ck);
    cv = wmma_bf16(a.b, bv.b, cv);
  }

  // C/D layout: elem r -> row (r + 8*hl), col n
  #pragma unroll
  for (int r = 0; r < 8; ++r) {
    const int srow = row_tile * 16 + r + 8 * hl;
    Qb[(size_t)srow * DIM + e0 + n] = (__bf16)cq[r];
    Vf[(size_t)srow * DIM + e0 + n] = cv[r];
  }
  // K: scatter into packed B-operand layout for the scores GEMM
  const int b     = row_tile >> 8;           // 256 row-tiles per batch
  const int kt    = row_tile & 255;
  const int dgk   = col_tile >> 1;
  const int halfF = col_tile & 1;
  __bf16* kbase = Ktp + ((size_t)(b * 256 + kt) * 8 + dgk) * 512;
  #pragma unroll
  for (int r = 0; r < 8; ++r) {
    const int lanep = halfF * 16 + r + 8 * hl;
    kbase[lanep * 16 + n] = (__bf16)ck[r];
  }
}

// ---------------------------------------------------------------------------
// Kernel 2: attention.  Workgroup = 8 waves, each wave owns 16 query rows.
// Key tiles are staged global->LDS with CDNA5 async loads (ASYNCcnt),
// double-buffered, loaded ONCE per workgroup and consumed by all 8 waves via
// ds_load_b128.  Sweep 1: online softmax stats; sweep 2: per-key prob sums
// into private LDS slices (deterministic, no atomics).
// LDS: 128KB accumulators + 2*32KB K staging = 192KB.
// ---------------------------------------------------------------------------
__global__ __launch_bounds__(256) void attn_kernel(
    const __bf16* __restrict__ Qb,
    const __bf16* __restrict__ Ktp,
    float* __restrict__ part)
{
  extern __shared__ char smem[];
  float* acc   = (float*)smem;               // 8 waves * 4096 f32 = 128KB
  char*  kst   = smem + 131072;              // 2 x 32KB staging
  const int tid  = threadIdx.x;
  const int w    = tid >> 5;
  const int lane = tid & 31;
  const int hl   = lane >> 4;
  const int n    = lane & 15;
  const int bb    = blockIdx.x >> 5;         // batch
  const int chunk = blockIdx.x & 31;         // 128-query chunk
  const int q0    = chunk * 128 + w * 16;

  float* myacc = acc + w * 4096;
  for (int i = lane; i < 4096; i += 32) myacc[i] = 0.0f;

  // Load this wave's 16x256 Q block as 8 resident A-tiles.
  v16bf qa[8];
  const __bf16* qrow = Qb + (size_t)(bb * SEQ + q0 + n) * DIM;
  #pragma unroll
  for (int dg = 0; dg < 8; ++dg) {
    const int d0 = dg * 32;
    us8 lo = *(const us8*)(qrow + d0 + 16 * hl);
    us8 hi = *(const us8*)(qrow + d0 + 16 * hl + 8);
    BfPack u;
    #pragma unroll
    for (int i = 0; i < 8; ++i) { u.u[i] = lo[i]; u.u[i + 8] = hi[i]; }
    qa[dg] = u.b;
  }

  const __bf16* kb_b = Ktp + (size_t)bb * (256u * 8u * 512u);
  const float sc = 0.0625f;                  // 1/sqrt(256)

  // Cooperative async copy of one 32KB key-group (64 keys) into LDS buffer.
  auto issue_group = [&](int buf, int g) {
    unsigned long long ga =
        (unsigned long long)(uintptr_t)((const char*)kb_b + (size_t)g * 32768 + tid * 128);
    unsigned la = (unsigned)(uintptr_t)(kst + buf * 32768 + tid * 128);
    #pragma unroll
    for (int i = 0; i < 8; ++i) {
      asm volatile("global_load_async_to_lds_b128 %0, %1, off"
                   :: "v"(la + i * 16), "v"(ga + (unsigned long long)(i * 16))
                   : "memory");
    }
  };

  float m8[8], l8[8];
  #pragma unroll
  for (int r = 0; r < 8; ++r) { m8[r] = -3.0e38f; l8[r] = 0.0f; }

  // ---- Sweep 1: online softmax stats, 64 keys per iteration ----
  issue_group(0, 0);
  for (int g = 0; g < 64; ++g) {
    const int cur = g & 1;
    asm volatile("s_wait_asynccnt 0x0" ::: "memory");
    __syncthreads();                          // staged data visible to all waves
    if (g < 63) issue_group(cur ^ 1, g + 1);  // prefetch next group
    const char* kb = kst + cur * 32768 + lane * 32;
    v8f c[4];
    #pragma unroll
    for (int t = 0; t < 4; ++t) c[t] = (v8f){0.f,0.f,0.f,0.f,0.f,0.f,0.f,0.f};
    #pragma unroll
    for (int dg = 0; dg < 8; ++dg) {
      #pragma unroll
      for (int t = 0; t < 4; ++t) {
        BfPack u; u.u = *(const us16*)(kb + t * 8192 + dg * 1024);
        c[t] = wmma_bf16(qa[dg], u.b, c[t]);
      }
    }
    #pragma unroll
    for (int r = 0; r < 8; ++r) {
      const float s0 = c[0][r] * sc, s1 = c[1][r] * sc;
      const float s2 = c[2][r] * sc, s3 = c[3][r] * sc;
      float v = fmaxf(fmaxf(s0, s1), fmaxf(s2, s3));
      v = fmaxf(v, __shfl_xor(v, 1));
      v = fmaxf(v, __shfl_xor(v, 2));
      v = fmaxf(v, __shfl_xor(v, 4));
      v = fmaxf(v, __shfl_xor(v, 8));
      const float mnew = fmaxf(m8[r], v);
      float su = __expf(s0 - mnew) + __expf(s1 - mnew) +
                 __expf(s2 - mnew) + __expf(s3 - mnew);
      su += __shfl_xor(su, 1);
      su += __shfl_xor(su, 2);
      su += __shfl_xor(su, 4);
      su += __shfl_xor(su, 8);
      l8[r] = l8[r] * __expf(m8[r] - mnew) + su;
      m8[r] = mnew;
    }
    __syncthreads();                          // all waves done with buffer cur
  }

  float il[8];
  #pragma unroll
  for (int r = 0; r < 8; ++r) il[r] = 1.0f / l8[r];

  // ---- Sweep 2: accumulate per-key probability sums over this wave's 16 q ----
  issue_group(0, 0);
  for (int g = 0; g < 64; ++g) {
    const int cur = g & 1;
    asm volatile("s_wait_asynccnt 0x0" ::: "memory");
    __syncthreads();
    if (g < 63) issue_group(cur ^ 1, g + 1);
    const char* kb = kst + cur * 32768 + lane * 32;
    v8f c[4];
    #pragma unroll
    for (int t = 0; t < 4; ++t) c[t] = (v8f){0.f,0.f,0.f,0.f,0.f,0.f,0.f,0.f};
    #pragma unroll
    for (int dg = 0; dg < 8; ++dg) {
      #pragma unroll
      for (int t = 0; t < 4; ++t) {
        BfPack u; u.u = *(const us16*)(kb + t * 8192 + dg * 1024);
        c[t] = wmma_bf16(qa[dg], u.b, c[t]);
      }
    }
    #pragma unroll
    for (int t = 0; t < 4; ++t) {
      float ks = 0.0f;
      #pragma unroll
      for (int r = 0; r < 8; ++r)
        ks += __expf(c[t][r] * sc - m8[r]) * il[r];
      ks += __shfl_xor(ks, 16);              // combine the two row-halves
      if (hl == 0) myacc[g * 64 + t * 16 + n] += ks;  // distinct addr per lane
    }
    __syncthreads();
  }

  // Fixed-order intra-workgroup reduction -> deterministic partials
  float* pr = part + (size_t)(bb * 32 + chunk) * 4096;
  for (int i = tid; i < 4096; i += 256) {
    float s = 0.0f;
    #pragma unroll
    for (int ww = 0; ww < 8; ++ww) s += acc[ww * 4096 + i];
    pr[i] = s;
  }
}

// ---------------------------------------------------------------------------
// Kernel 3: fixed-order reduction of the 32 query-chunk partials per batch,
// scale by 1/S -> final attn output.
// ---------------------------------------------------------------------------
__global__ __launch_bounds__(256) void reduce_attn_kernel(
    const float* __restrict__ part, float* __restrict__ attn)
{
  const int idx = blockIdx.x * 256 + threadIdx.x;   // 0..32767
  const int b = idx >> 12;
  const int k = idx & 4095;
  const float* p = part + (size_t)b * 32 * 4096 + k;
  float s = 0.0f;
  #pragma unroll
  for (int c = 0; c < 32; ++c) s += p[(size_t)c * 4096];
  attn[idx] = s * (1.0f / (float)SEQ);
}

// ---------------------------------------------------------------------------
// Kernel 4: context[b][e] = sum_k attn[b][k] * V[b][k][e]   (tiny GEMV)
// ---------------------------------------------------------------------------
__global__ __launch_bounds__(256) void context_kernel(
    const float* __restrict__ attn, const float* __restrict__ Vf,
    float* __restrict__ ctx)
{
  const int idx = blockIdx.x * 256 + threadIdx.x;   // 0..2047
  const int b = idx >> 8;
  const int e = idx & 255;
  const float* a = attn + b * SEQ;
  const float* v = Vf + (size_t)b * SEQ * DIM + e;
  float s = 0.0f;
  for (int k = 0; k < SEQ; k += 4) {
    s += a[k]     * v[(size_t)k * DIM];
    s += a[k + 1] * v[(size_t)(k + 1) * DIM];
    s += a[k + 2] * v[(size_t)(k + 2) * DIM];
    s += a[k + 3] * v[(size_t)(k + 3) * DIM];
  }
  ctx[idx] = s;
}

// ---------------------------------------------------------------------------
// Launch.  Workspace layout (bytes):
//   [ 0,16MB)   Sb   bf16 src copy (row-major)
//   [16,32MB)   Qb   bf16 row-major
//   [32,48MB)   Ktp  bf16 packed B-operand
//   [48,49MB)   Wp   bf16 packed weights (3 * 128KB)
//   [49,81MB)   Vf   f32 row-major
//   [81,85MB)   part f32 per-chunk attn partials
// ---------------------------------------------------------------------------
extern "C" void kernel_launch(void* const* d_in, const int* in_sizes, int n_in,
                              void* d_out, int out_size, void* d_ws, size_t ws_size,
                              hipStream_t stream) {
  (void)in_sizes; (void)n_in; (void)out_size; (void)ws_size;
  const float* src = (const float*)d_in[0];
  const float* Wq  = (const float*)d_in[1];
  const float* Wk  = (const float*)d_in[2];
  const float* Wv  = (const float*)d_in[3];
  float* out = (float*)d_out;                     // [0,2048) context, [2048,34816) attn
  char*  ws  = (char*)d_ws;

  __bf16* Sb   = (__bf16*)(ws);
  __bf16* Qb   = (__bf16*)(ws + (size_t)(16u << 20));
  __bf16* Ktp  = (__bf16*)(ws + (size_t)(32u << 20));
  __bf16* Wp   = (__bf16*)(ws + (size_t)(48u << 20));
  float*  Vf   = (float*) (ws + (size_t)(49u << 20));
  float*  part = (float*) (ws + (size_t)(81u << 20));

  convert_src_kernel<<<4096, 256, 0, stream>>>(src, Sb);
  convert_w_kernel<<<768, 256, 0, stream>>>(Wq, Wk, Wv, Wp);
  qkv_proj_kernel<<<4096, 256, 0, stream>>>(Sb, Wp, Qb, Ktp, Vf);
  attn_kernel<<<BATCH * (SEQ / 128), 256, 196608, stream>>>(Qb, Ktp, part);
  reduce_attn_kernel<<<128, 256, 0, stream>>>(part, out + BATCH * DIM);
  context_kernel<<<8, 256, 0, stream>>>(out + BATCH * DIM, Vf, out);
}
